// GATNet_66726611911377
// MI455X (gfx1250) — matrix operand order repaired and checked by reading, not compiled, not run on hardware
//
#include <hip/hip_runtime.h>
#include <math.h>

typedef __attribute__((ext_vector_type(16))) _Float16 v16h;
typedef __attribute__((ext_vector_type(8)))  _Float16 v8h;
typedef __attribute__((ext_vector_type(8)))  float    v8f;

// ============================================================================
// fp32 -> fp16 elementwise convert (RTE), coalesced.
// ============================================================================
__global__ __launch_bounds__(256)
void cvt_f32_to_f16(const float* __restrict__ in, _Float16* __restrict__ out,
                    long long n)
{
    long long t = blockIdx.x * (long long)blockDim.x + threadIdx.x;
    if (t >= n) return;
    out[t] = (_Float16)in[t];
}

// ============================================================================
// GEMM:  C[M,N] = A[M,K] * B[N,K]^T   (A,B fp16 row-major; C fp32; f32 acc)
// One wave32 computes a 16 x (16*NT) strip of C.  With f16 operands in memory
// each fragment is exactly two b128 loads straight into the WMMA source
// registers -- no converts in the loop, so loads clause up and drain with
// descending waitcnts ahead of the NT back-to-back WMMAs.
//   A 16x32 f16 frag (row = lane&15): elems 0..7  -> K = kb + half*8 + j
//                                     elems 8..15 -> K = kb + 16 + half*8 + j
//   B 32x16 f16 frag (col = lane&15): same K mapping as A
// Requires M%16==0, N%(16*NT)==0, K%32==0.
// ============================================================================
template <int NT>
__global__ __launch_bounds__(128)
void gemm_nt_wmma_f16(const _Float16* __restrict__ A, const _Float16* __restrict__ B,
                      float* __restrict__ C, int M, int N, int K)
{
    const int groupsN = N / (16 * NT);
    const int total   = (M >> 4) * groupsN;
    const int wave    = blockIdx.x * blockDim.y + threadIdx.y;
    if (wave >= total) return;                    // uniform per wave -> EXEC all-1s

    const int tm   = wave / groupsN;
    const int tg   = wave % groupsN;
    const int lane = threadIdx.x;                 // 0..31
    const int half = lane >> 4;
    const int idx  = lane & 15;

    const _Float16* arow = A + (size_t)(tm * 16 + idx) * K;
    const _Float16* brow[NT];
    #pragma unroll
    for (int j = 0; j < NT; ++j)
        brow[j] = B + (size_t)((tg * NT + j) * 16 + idx) * K;

    v8f acc[NT];
    #pragma unroll
    for (int j = 0; j < NT; ++j) acc[j] = (v8f){};

    for (int kb = 0; kb < K; kb += 32) {
        // ---- all fragment loads for this K-step: 2 + 2*NT b128 loads ------
        v8h a0 = *(const v8h*)(arow + kb + half * 8);
        v8h a1 = *(const v8h*)(arow + kb + 16 + half * 8);
        v8h b0[NT], b1[NT];
        #pragma unroll
        for (int j = 0; j < NT; ++j) {
            b0[j] = *(const v8h*)(brow[j] + kb + half * 8);
            b1[j] = *(const v8h*)(brow[j] + kb + 16 + half * 8);
        }

        v16h a = __builtin_shufflevector(a0, a1,
                     0,1,2,3,4,5,6,7,8,9,10,11,12,13,14,15);
        #pragma unroll
        for (int j = 0; j < NT; ++j) {
            v16h b = __builtin_shufflevector(b0[j], b1[j],
                         0,1,2,3,4,5,6,7,8,9,10,11,12,13,14,15);
            acc[j] = __builtin_amdgcn_wmma_f32_16x16x32_f16(
                         false, a, false, b, (short)0, acc[j], false, false);
        }
    }

    // D layout: VGPR v, lane -> row = tm*16 + 8*half + v, col = tile_n*16 + idx
    #pragma unroll
    for (int j = 0; j < NT; ++j) {
        float* crow = C + (size_t)(tm * 16 + half * 8) * N
                        + (size_t)((tg * NT + j) * 16 + idx);
        #pragma unroll
        for (int v = 0; v < 8; ++v) crow[(size_t)v * N] = acc[j][v];
    }
}

// ============================================================================
// Per-node attention scores: s_src[i] = h[i]·att_src ; s_dst[i] = h[i]·att_dst
// One wave32 per node, cross-lane reduce with shuffles.
// ============================================================================
__global__ __launch_bounds__(256)
void scores_kernel(const float* __restrict__ h, const float* __restrict__ att_src,
                   const float* __restrict__ att_dst, float* __restrict__ s_src,
                   float* __restrict__ s_dst, int N, int D)
{
    int gwave = (blockIdx.x * blockDim.x + threadIdx.x) >> 5;
    int lane  = threadIdx.x & 31;
    if (gwave >= N) return;
    const float* row = h + (size_t)gwave * D;
    float a = 0.f, b = 0.f;
    for (int d = lane; d < D; d += 32) {
        float v = row[d];
        a += v * att_src[d];
        b += v * att_dst[d];
    }
    #pragma unroll
    for (int off = 16; off > 0; off >>= 1) {
        a += __shfl_down(a, off, 32);
        b += __shfl_down(b, off, 32);
    }
    if (lane == 0) { s_src[gwave] = a; s_dst[gwave] = b; }
}

// ---- order-preserving float<->uint encoding for atomic segment-max ---------
__device__ __forceinline__ unsigned fenc(float f) {
    unsigned u = __float_as_uint(f);
    return (u >> 31) ? ~u : (u | 0x80000000u);
}
__device__ __forceinline__ float fdec(unsigned u) {
    return __uint_as_float((u >> 31) ? (u ^ 0x80000000u) : ~u);
}

// Pass 1: e = leaky_relu(s_src[row] + s_dst[col]); segment max via atomicMax.
__global__ __launch_bounds__(256)
void edge_logits(const int* __restrict__ row, const int* __restrict__ col,
                 const float* __restrict__ ssrc, const float* __restrict__ sdst,
                 float* __restrict__ e, unsigned* __restrict__ menc, int E)
{
    int t = blockIdx.x * blockDim.x + threadIdx.x;
    if (t >= E) return;
    int r = row[t], c = col[t];
    float v = ssrc[r] + sdst[c];
    v = (v >= 0.f) ? v : 0.2f * v;
    e[t] = v;
    atomicMax(menc + r, fenc(v));
}

// Pass 2: ex = exp(e - m[row]); segment sum via atomicAdd. (in-place on e)
__global__ __launch_bounds__(256)
void edge_exp(const int* __restrict__ row, float* __restrict__ e,
              const unsigned* __restrict__ menc, float* __restrict__ denom, int E)
{
    int t = blockIdx.x * blockDim.x + threadIdx.x;
    if (t >= E) return;
    int r = row[t];
    float x = expf(e[t] - fdec(menc[r]));
    e[t] = x;
    atomicAdd(denom + r, x);
}

// Pass 3: finalize softmax weights per edge (in-place: ex -> alpha).
__global__ __launch_bounds__(256)
void edge_alpha(const int* __restrict__ row, float* __restrict__ e,
                const float* __restrict__ denom, int E)
{
    int t = blockIdx.x * blockDim.x + threadIdx.x;
    if (t >= E) return;
    e[t] = e[t] / denom[row[t]];
}

// Pass 4 (SpMM scatter): out[row] += alpha * feats[col]
// CH = D/4 threads per edge, each handles one float4 chunk (chunk fastest ->
// each edge's feature row is read 128B-contiguously by a sub-wave).
template <int CH>
__global__ __launch_bounds__(256)
void edge_spmm_vec(const int* __restrict__ row, const int* __restrict__ col,
                   const float* __restrict__ alpha,
                   const float* __restrict__ feats, float* __restrict__ out, int E)
{
    unsigned t     = blockIdx.x * blockDim.x + threadIdx.x;
    unsigned edge  = t / CH;
    unsigned chunk = t % CH;
    if (edge >= (unsigned)E) return;
    int r = row[edge], c = col[edge];
    float al = alpha[edge];
    const int D = CH * 4;
    const float4 f = *(const float4*)(feats + (size_t)c * D + chunk * 4);
    float* o = out + (size_t)r * D + chunk * 4;
    atomicAdd(o + 0, al * f.x);
    atomicAdd(o + 1, al * f.y);
    atomicAdd(o + 2, al * f.z);
    atomicAdd(o + 3, al * f.w);
}

// Fallback for arbitrary D.
__global__ __launch_bounds__(256)
void edge_spmm_generic(const int* __restrict__ row, const int* __restrict__ col,
                       const float* __restrict__ alpha,
                       const float* __restrict__ feats, float* __restrict__ out,
                       int E, int D)
{
    int t = blockIdx.x * blockDim.x + threadIdx.x;
    if (t >= E) return;
    int r = row[t], c = col[t];
    float al = alpha[t];
    const float* f = feats + (size_t)c * D;
    float* o = out + (size_t)r * D;
    for (int d = 0; d < D; ++d) atomicAdd(o + d, al * f[d]);
}

__global__ __launch_bounds__(256)
void elu_kernel(float* __restrict__ x, long long n)
{
    long long t = blockIdx.x * (long long)blockDim.x + threadIdx.x;
    if (t >= n) return;
    float v = x[t];
    x[t] = (v > 0.f) ? v : expm1f(v);
}

// ============================================================================
static inline void launch_cvt(const float* in, _Float16* out, long long n,
                              hipStream_t stream)
{
    cvt_f32_to_f16<<<(unsigned)((n + 255) / 256), 256, 0, stream>>>(in, out, n);
}

static inline void launch_spmm(const int* row, const int* col, const float* alpha,
                               const float* feats, float* out,
                               int E, int D, hipStream_t stream)
{
    if (D == 128) {
        long long th = (long long)E * 32;
        edge_spmm_vec<32><<<(unsigned)((th + 255) / 256), 256, 0, stream>>>(row, col, alpha, feats, out, E);
    } else if (D == 64) {
        long long th = (long long)E * 16;
        edge_spmm_vec<16><<<(unsigned)((th + 255) / 256), 256, 0, stream>>>(row, col, alpha, feats, out, E);
    } else {
        edge_spmm_generic<<<(E + 255) / 256, 256, 0, stream>>>(row, col, alpha, feats, out, E, D);
    }
}

// Convert A,B to f16 pools, then run the WMMA GEMM.
static inline void launch_gemm(const float* A, const float* B, float* C,
                               _Float16* Ah, _Float16* Bh,
                               int M, int N, int K, hipStream_t stream)
{
    launch_cvt(A, Ah, (long long)M * K, stream);
    launch_cvt(B, Bh, (long long)N * K, stream);
    dim3 wblock(32, 4);          // 4 waves per block
    if (N % 64 == 0) {
        int waves = (M / 16) * (N / 64);
        gemm_nt_wmma_f16<4><<<(waves + 3) / 4, wblock, 0, stream>>>(Ah, Bh, C, M, N, K);
    } else {
        int waves = (M / 16) * (N / 16);
        gemm_nt_wmma_f16<1><<<(waves + 3) / 4, wblock, 0, stream>>>(Ah, Bh, C, M, N, K);
    }
}

static inline void gat_layer(const float* feats, const int* row, const int* col,
                             const float* asrc, const float* adst,
                             float* e, unsigned* menc, float* denom,
                             float* ssrc, float* sdst, float* out,
                             int N, int E, int D, hipStream_t stream)
{
    scores_kernel<<<(N + 7) / 8, 256, 0, stream>>>(feats, asrc, adst, ssrc, sdst, N, D);
    hipMemsetAsync(menc,  0, (size_t)N * sizeof(unsigned), stream);
    hipMemsetAsync(denom, 0, (size_t)N * sizeof(float),    stream);
    int eb = (E + 255) / 256;
    edge_logits<<<eb, 256, 0, stream>>>(row, col, ssrc, sdst, e, menc, E);
    edge_exp   <<<eb, 256, 0, stream>>>(row, e, menc, denom, E);
    edge_alpha <<<eb, 256, 0, stream>>>(row, e, denom, E);
    hipMemsetAsync(out, 0, (size_t)N * D * sizeof(float), stream);
    launch_spmm(row, col, e, feats, out, E, D, stream);
}

extern "C" void kernel_launch(void* const* d_in, const int* in_sizes, int n_in,
                              void* d_out, int out_size, void* d_ws, size_t ws_size,
                              hipStream_t stream)
{
    const float* x   = (const float*)d_in[0];
    const int*   ei  = (const int*)  d_in[1];
    const float* W1  = (const float*)d_in[2];
    const float* W2  = (const float*)d_in[3];
    const float* as1 = (const float*)d_in[4];
    const float* ad1 = (const float*)d_in[5];
    const float* as2 = (const float*)d_in[6];
    const float* ad2 = (const float*)d_in[7];

    const int H   = in_sizes[4];               // 128
    const int OUT = in_sizes[6];               // 64
    const int IN  = in_sizes[2] / H;           // 256
    const int N   = in_sizes[0] / IN;          // 100000
    const int E   = in_sizes[1] / 2;           // 1600000

    const int* row = ei;                        // edge_index[0, :]
    const int* col = ei + E;                    // edge_index[1, :]

    // ---- workspace carve-out (256B aligned slices) -------------------------
    char*  ws  = (char*)d_ws;
    size_t off = 0;
    auto carve = [&](size_t bytes) -> char* {
        char* p = ws + off;
        off += (bytes + 255) & ~(size_t)255;
        return p;
    };
    float*    h1    = (float*)   carve((size_t)N * H * sizeof(float)); // h1, later h2
    float*    hatt  = (float*)   carve((size_t)N * H * sizeof(float)); // attn out / elu
    float*    ebuf  = (float*)   carve((size_t)E * sizeof(float));
    unsigned* menc  = (unsigned*)carve((size_t)N * sizeof(unsigned));
    float*    denom = (float*)   carve((size_t)N * sizeof(float));
    float*    ssrc  = (float*)   carve((size_t)N * sizeof(float));
    float*    sdst  = (float*)   carve((size_t)N * sizeof(float));
    // f16 pools, reused across the two GEMMs
    size_t aPoolElems = (size_t)N * ((IN > H) ? IN : H);
    size_t bPoolElems = (size_t)H * ((IN > OUT) ? IN : OUT);
    _Float16* Ah = (_Float16*)carve(aPoolElems * sizeof(_Float16));
    _Float16* Bh = (_Float16*)carve(bPoolElems * sizeof(_Float16));
    (void)ws_size; (void)n_in; (void)out_size;

    // ---- layer 1: h1 = x @ W1^T (WMMA), GAT attention, ELU ----------------
    launch_gemm(x, W1, h1, Ah, Bh, N, H, IN, stream);
    gat_layer(h1, row, col, as1, ad1, ebuf, menc, denom, ssrc, sdst, hatt,
              N, E, H, stream);
    elu_kernel<<<(unsigned)(((long long)N * H + 255) / 256), 256, 0, stream>>>(
        hatt, (long long)N * H);

    // ---- layer 2: h2 = hatt @ W2^T (WMMA), GAT attention -> d_out ---------
    float* h2 = h1;  // reuse
    launch_gemm(hatt, W2, h2, Ah, Bh, N, OUT, H, stream);
    gat_layer(h2, row, col, as2, ad2, ebuf, menc, denom, ssrc, sdst,
              (float*)d_out, N, E, OUT, stream);
}
